// Multihead_7172595384730
// MI455X (gfx1250) — compile-verified, matrix operand order checked by reading, not run on hardware
//
#include <hip/hip_runtime.h>
#include <hip/hip_bf16.h>
#include <math.h>

// ---------------- CDNA5 WMMA types ----------------
typedef __attribute__((ext_vector_type(16))) __bf16 v16bf;
typedef __attribute__((ext_vector_type(8)))  float  v8f;

#define DIMS   1024
#define HEADS  16
#define HD     64
#define CTX    2048
#define BATCH  2
#define MROWS  (BATCH * CTX)            // 4096
#define QKSCALE 0.3535533905932738f     // 64^-0.25

// LDS tiles: K-step 64 halves per row, padded to 72 halves = 36 dwords = 144B (16B-aligned rows)
#define LDSW  36                         // dwords per row
#define ATILE (64 * LDSW)                // 2304 dwords
#define BTILE (128 * LDSW)               // 4608 dwords

static __device__ __forceinline__ unsigned short f2bf(float f) {
    unsigned int u = __float_as_uint(f);
    u += 0x7FFFu + ((u >> 16) & 1u);    // round-to-nearest-even
    return (unsigned short)(u >> 16);
}

union FragBF { unsigned int u[8]; v16bf v; };

// Async global->LDS 16-byte copy (GLOBAL_LOAD_ASYNC_TO_LDS_B128, ASYNCcnt).
static __device__ __forceinline__ void asyncCopy16(unsigned int ldsAddr,
                                                   const void* gaddr) {
    asm volatile("global_load_async_to_lds_b128 %0, %1, off"
                 :: "v"(ldsAddr), "v"((unsigned long long)gaddr)
                 : "memory");
}
static __device__ __forceinline__ void asyncWait0() {
    asm volatile("s_wait_asynccnt 0x0" ::: "memory");
}
// Low 32 bits of a flat shared-aperture pointer == LDS byte offset (runtime cast).
static __device__ __forceinline__ unsigned int ldsAddrOf(const void* p) {
    return (unsigned int)(unsigned long long)p;
}

// Load one 16x32 bf16 WMMA fragment (A-layout; B symmetric with row=column)
// from an LDS tile with dword row stride LDSW. kHalf = 0 or 32 selects K sub-step.
// ISA layout: lane L holds row L%16; VGPR v holds K pair
//   k = (v>=4 ? 16 : 0) + (L>=16 ? 8 : 0) + (v&3)*2
static __device__ __forceinline__ v16bf ldsFrag(const unsigned int* lds32,
                                                int rowBase, int kHalf, int lane) {
    const int r  = rowBase + (lane & 15);
    const int hh = (lane >> 4) << 3;        // 0 or 8
    const int base = r * LDSW + (kHalf >> 1);
    FragBF f;
#pragma unroll
    for (int v = 0; v < 8; ++v) {
        const int k = ((v & 4) ? 16 : 0) + hh + ((v & 3) << 1);  // even
        f.u[v] = lds32[base + (k >> 1)];
    }
    return f.v;
}

// ---------------- fp32 -> bf16 conversion ----------------
__global__ void cvt_f32_bf16(const float* __restrict__ in,
                             unsigned short* __restrict__ out, int n) {
    int i = blockIdx.x * blockDim.x + threadIdx.x;
    const int stride = gridDim.x * blockDim.x;
    for (; i < n; i += stride) out[i] = f2bf(in[i]);
}

// ---------------- generic tiled WMMA GEMM: C = A(MxK) @ B(NxK)^T ----------------
// Block tile 64(M) x 128(N), 128 threads = 4 waves. Wave w owns N-subtiles
// {w*16, 64+w*16} x all 4 M-subtiles -> 8 accumulators, 16 WMMA per K-64 phase.
// Double-buffered LDS fed by GLOBAL_LOAD_ASYNC_TO_LDS_B128 (ASYNCcnt).
__global__ __launch_bounds__(128)
void gemm_wmma_bf16(const unsigned short* __restrict__ A, int lda,
                    long long aStrB, long long aStrH,
                    const unsigned short* __restrict__ B, int ldb,
                    long long bStrB, long long bStrH,
                    const float* __restrict__ bias,
                    const float* __restrict__ colScale,
                    long long csStrB, long long csStrH,
                    float outScale,
                    unsigned short* __restrict__ outBf,
                    float* __restrict__ outF32, int ldc,
                    long long cStrB, long long cStrH,
                    unsigned short* __restrict__ vTout,
                    float* __restrict__ zeroFlags,
                    const float* __restrict__ factorPtr,
                    int M, int N, int Kd)
{
    __shared__ unsigned int lds[2 * ATILE + 2 * BTILE];   // 55296 B
    const unsigned int ldsBase = ldsAddrOf(lds);          // runtime, not static init

    const int tid  = threadIdx.x;
    const int lane = tid & 31;
    const int wave = tid >> 5;
    const int z  = blockIdx.z;
    const int zb = z >> 4, zh = z & 15;
    const long long aOff = (long long)zb * aStrB + (long long)zh * aStrH;
    const long long bOff = (long long)zb * bStrB + (long long)zh * bStrH;
    const int mBase = blockIdx.y * 64;
    const int nBase = blockIdx.x * 128;

    // per-thread staging coordinates (A: 512 uint4, B: 1024 uint4 per tile)
    const int sr = tid >> 3, sq = tid & 7;

    auto stageAsync = [&](int buf, int kb) {
        const unsigned int aB = ldsBase + (unsigned int)(buf * ATILE) * 4u;
        const unsigned int bB = ldsBase + (unsigned int)(2 * ATILE + buf * BTILE) * 4u;
#pragma unroll
        for (int it = 0; it < 4; ++it) {                  // A: 64 rows
            const int r = sr + it * 16;
            asyncCopy16(aB + (unsigned int)((r * LDSW + (sq << 2)) << 2),
                        A + aOff + (long long)(mBase + r) * lda + kb + (sq << 3));
        }
#pragma unroll
        for (int it = 0; it < 8; ++it) {                  // B: 128 rows
            const int r = sr + it * 16;
            asyncCopy16(bB + (unsigned int)((r * LDSW + (sq << 2)) << 2),
                        B + bOff + (long long)(nBase + r) * ldb + kb + (sq << 3));
        }
    };

    v8f acc[2][4] = {};

    stageAsync(0, 0);
    int cur = 0;
    for (int kb = 0; kb < Kd; kb += 64) {
        asyncWait0();          // this wave's async tile is in LDS
        __syncthreads();       // all waves' tiles visible; prev compute done
        if (kb + 64 < Kd) stageAsync(cur ^ 1, kb + 64);   // overlap next tile
        const unsigned int* cA = lds + cur * ATILE;
        const unsigned int* cB = lds + 2 * ATILE + cur * BTILE;
#pragma unroll
        for (int ks = 0; ks < 2; ++ks) {
            const int kh = ks << 5;
            const v16bf b0 = ldsFrag(cB, (wave << 4), kh, lane);
            const v16bf b1 = ldsFrag(cB, 64 + (wave << 4), kh, lane);
#pragma unroll
            for (int ms = 0; ms < 4; ++ms) {
                const v16bf af = ldsFrag(cA, ms << 4, kh, lane);
                acc[0][ms] = __builtin_amdgcn_wmma_f32_16x16x32_bf16(
                    false, af, false, b0, (short)0, acc[0][ms], false, false);
                acc[1][ms] = __builtin_amdgcn_wmma_f32_16x16x32_bf16(
                    false, af, false, b1, (short)0, acc[1][ms], false, false);
            }
        }
        cur ^= 1;
    }

    // ---------------- epilogue ----------------
    float zf = 0.0f;
    if (zeroFlags) {   // clipped softplus(factor)
        const float sp = log1pf(__expf(factorPtr[0]));
        zf = fminf(fmaxf(sp, 0.0f), 0.001f);
    }
#pragma unroll
    for (int g = 0; g < 2; ++g) {
        const int n = nBase + g * 64 + (wave << 4) + (lane & 15);
        const float cs   = colScale ? colScale[(long long)zb * csStrB +
                                               (long long)zh * csStrH + n] : 1.0f;
        const float bval = bias ? bias[n] : 0.0f;
#pragma unroll
        for (int ms = 0; ms < 4; ++ms) {
#pragma unroll
            for (int vi = 0; vi < 8; ++vi) {
                const int m = mBase + (ms << 4) + vi + ((lane >> 4) << 3);
                const float raw = acc[g][ms][vi];
                const float val = (raw + bval) * outScale * cs;
                if (vTout) {   // V projection: head-transposed [(b*16+h)*64+c][t]
                    const int b = m >> 11, t = m & 2047, h = n >> 6, c = n & 63;
                    vTout[(((long long)(b * HEADS + h) * HD + c) << 11) + t] = f2bf(val);
                } else if (outBf) {
                    outBf[(long long)m * ldc + n] = f2bf(val);
                } else {
                    outF32[(long long)zb * cStrB + (long long)zh * cStrH +
                           (long long)m * ldc + n] = val;
                }
                if (zeroFlags && ((n & 63) == 0)) {   // first feature of a head
                    const int b = m >> 11, t = m & 2047, h = n >> 6;
                    zeroFlags[((long long)(b * HEADS + h) << 11) + t] =
                        (raw == 0.0f) ? zf : 1.0f;
                }
            }
        }
    }
}

// ---------------- softmax row statistics over qk (2048-wide rows) ----------------
__global__ __launch_bounds__(256)
void softmax_stats(const float* __restrict__ qk,
                   float* __restrict__ rowMax, float* __restrict__ rowRinv) {
    __shared__ float red[256];
    const long long row = blockIdx.x;
    const float* p = qk + (row << 11);
    const int tid = threadIdx.x;
    float mx = -3.4e38f;
    for (int i = tid; i < CTX; i += 256) mx = fmaxf(mx, p[i]);
    red[tid] = mx; __syncthreads();
    for (int s = 128; s > 0; s >>= 1) {
        if (tid < s) red[tid] = fmaxf(red[tid], red[tid + s]);
        __syncthreads();
    }
    mx = red[0]; __syncthreads();
    float sum = 0.f;
    for (int i = tid; i < CTX; i += 256) sum += __expf(p[i] - mx);
    red[tid] = sum; __syncthreads();
    for (int s = 128; s > 0; s >>= 1) {
        if (tid < s) red[tid] += red[tid + s];
        __syncthreads();
    }
    if (tid == 0) { rowMax[row] = mx; rowRinv[row] = 1.0f / red[0]; }
}

// ---------------- wv = softmax(qk) @ V  (probabilities built on the fly) ----------------
// Block tile 64(M) x 64(N=HD), K-step 64 over CTX -> 8 WMMA per phase per wave.
// Probability tile converted through VALU (exp) into LDS; V tile staged async.
__global__ __launch_bounds__(128)
void attn_pv_wmma(const float* __restrict__ qk,
                  const float* __restrict__ rowMax,
                  const float* __restrict__ rowRinv,
                  const unsigned short* __restrict__ vT,
                  unsigned short* __restrict__ wvOut)
{
    __shared__ unsigned int lds[4 * ATILE];   // A0, A1, B0, B1 -> 36864 B
    __shared__ float sMax[64];
    __shared__ float sRinv[64];

    const int tid = threadIdx.x, lane = tid & 31, wave = tid >> 5;
    const int z = blockIdx.z;                    // b*16 + h
    const int mBase = blockIdx.x * 64;
    const long long qOff = (long long)z << 22;   // z * 2048 * 2048
    const long long sOff = (long long)z << 11;
    const long long vOff = (long long)z * (HD * CTX);
    const unsigned int ldsBase = ldsAddrOf(lds);

    if (tid < 64) {
        sMax[tid]  = rowMax[sOff + mBase + tid];
        sRinv[tid] = rowRinv[sOff + mBase + tid];
    }
    __syncthreads();

    const int sr = tid >> 3, sq = tid & 7;       // B staging coords

    auto stageB = [&](int buf, int kb) {
        const unsigned int bB = ldsBase + (unsigned int)((2 + buf) * ATILE) * 4u;
#pragma unroll
        for (int it = 0; it < 4; ++it) {                 // 64 rows of vT
            const int r = sr + it * 16;
            asyncCopy16(bB + (unsigned int)((r * LDSW + (sq << 2)) << 2),
                        vT + vOff + ((long long)r << 11) + kb + (sq << 3));
        }
    };
    auto stageA = [&](int buf, int kb) {
        unsigned int* dst = lds + buf * ATILE;
#pragma unroll
        for (int it = 0; it < 8; ++it) {                 // 64 rows x 16 float4
            const int i = tid + it * 128;
            const int r = i >> 4, c4 = i & 15;
            const float4 v = *(const float4*)
                (qk + qOff + ((long long)(mBase + r) << 11) + kb + (c4 << 2));
            const float mx = sMax[r], ri = sRinv[r];
            const unsigned int u01 = (unsigned int)f2bf(__expf(v.x - mx) * ri) |
                                     ((unsigned int)f2bf(__expf(v.y - mx) * ri) << 16);
            const unsigned int u23 = (unsigned int)f2bf(__expf(v.z - mx) * ri) |
                                     ((unsigned int)f2bf(__expf(v.w - mx) * ri) << 16);
            dst[r * LDSW + (c4 << 1)]     = u01;
            dst[r * LDSW + (c4 << 1) + 1] = u23;
        }
    };

    v8f acc[4] = {};

    stageB(0, 0);
    stageA(0, 0);
    int cur = 0;
    for (int kb = 0; kb < CTX; kb += 64) {
        asyncWait0();
        __syncthreads();
        if (kb + 64 < CTX) {
            stageB(cur ^ 1, kb + 64);
            stageA(cur ^ 1, kb + 64);
        }
        const unsigned int* cA = lds + cur * ATILE;
        const unsigned int* cB = lds + (2 + cur) * ATILE;
#pragma unroll
        for (int ks = 0; ks < 2; ++ks) {
            const int kh = ks << 5;
            const v16bf bf = ldsFrag(cB, wave << 4, kh, lane);
#pragma unroll
            for (int ms = 0; ms < 4; ++ms) {
                const v16bf af = ldsFrag(cA, ms << 4, kh, lane);
                acc[ms] = __builtin_amdgcn_wmma_f32_16x16x32_bf16(
                    false, af, false, bf, (short)0, acc[ms], false, false);
            }
        }
        cur ^= 1;
    }

    const int b = z >> 4, h = z & 15;
    const int n = (wave << 4) + (lane & 15);     // 0..63
    const int col = h * HD + n;
#pragma unroll
    for (int ms = 0; ms < 4; ++ms)
#pragma unroll
        for (int vi = 0; vi < 8; ++vi) {
            const int m = mBase + (ms << 4) + vi + ((lane >> 4) << 3);
            const long long row = (long long)(b * CTX + m);
            wvOut[row * DIMS + col] = f2bf(acc[ms][vi]);
        }
}

// ---------------- host orchestration ----------------
extern "C" void kernel_launch(void* const* d_in, const int* in_sizes, int n_in,
                              void* d_out, int out_size, void* d_ws, size_t ws_size,
                              hipStream_t stream) {
    const float* x      = (const float*)d_in[0];
    const float* Wq     = (const float*)d_in[1];
    const float* bq     = (const float*)d_in[2];
    const float* Wk     = (const float*)d_in[3];
    const float* Wv     = (const float*)d_in[4];
    const float* bv     = (const float*)d_in[5];
    const float* Wout   = (const float*)d_in[6];
    const float* bout   = (const float*)d_in[7];
    const float* factor = (const float*)d_in[8];

    float* outPtr = (float*)d_out;                       // (b, ctx, dims)
    float* qkOut  = outPtr + (long long)MROWS * DIMS;    // (b, h, ctx, ctx)

    // workspace carve-up (256B aligned)
    char* wsb = (char*)d_ws;
    size_t o = 0;
    auto carve = [&](size_t bytes) -> void* {
        void* p = wsb + o;
        o += (bytes + 255) & ~(size_t)255;
        return p;
    };
    unsigned short* xBf   = (unsigned short*)carve((size_t)MROWS * DIMS * 2);
    unsigned short* wqBf  = (unsigned short*)carve((size_t)DIMS * DIMS * 2);
    unsigned short* wkBf  = (unsigned short*)carve((size_t)DIMS * DIMS * 2);
    unsigned short* wvwBf = (unsigned short*)carve((size_t)DIMS * DIMS * 2);
    unsigned short* woBf  = (unsigned short*)carve((size_t)DIMS * DIMS * 2);
    unsigned short* Qbf   = (unsigned short*)carve((size_t)MROWS * DIMS * 2);
    unsigned short* Kbf   = (unsigned short*)carve((size_t)MROWS * DIMS * 2);
    unsigned short* vT    = (unsigned short*)carve((size_t)BATCH * HEADS * HD * CTX * 2);
    unsigned short* wvBf  = (unsigned short*)carve((size_t)MROWS * DIMS * 2);
    unsigned short* o1Bf  = (unsigned short*)carve((size_t)MROWS * DIMS * 2);
    float* zeroFlags      = (float*)carve((size_t)BATCH * HEADS * CTX * 4);
    float* rowMax         = (float*)carve((size_t)BATCH * HEADS * CTX * 4);
    float* rowRinv        = (float*)carve((size_t)BATCH * HEADS * CTX * 4);
    (void)ws_size; (void)n_in; (void)in_sizes; (void)out_size;

    // 1) fp32 -> bf16 conversions
    cvt_f32_bf16<<<2048, 256, 0, stream>>>(x,    xBf,   MROWS * DIMS);
    cvt_f32_bf16<<<1024, 256, 0, stream>>>(Wq,   wqBf,  DIMS * DIMS);
    cvt_f32_bf16<<<1024, 256, 0, stream>>>(Wk,   wkBf,  DIMS * DIMS);
    cvt_f32_bf16<<<1024, 256, 0, stream>>>(Wv,   wvwBf, DIMS * DIMS);
    cvt_f32_bf16<<<1024, 256, 0, stream>>>(Wout, woBf,  DIMS * DIMS);

    const dim3 blk(128);
    const dim3 gProj(DIMS / 128, MROWS / 64, 1);         // 8 x 64

    // 2) Q = (x@Wq^T + bq) * scale   -> bf16
    gemm_wmma_bf16<<<gProj, blk, 0, stream>>>(
        xBf, DIMS, 0, 0, wqBf, DIMS, 0, 0, bq,
        nullptr, 0, 0, QKSCALE,
        Qbf, nullptr, DIMS, 0, 0, nullptr, nullptr, nullptr,
        MROWS, DIMS, DIMS);
    // 3) K = (x@Wk^T) * scale        -> bf16, plus zero-feature flags
    gemm_wmma_bf16<<<gProj, blk, 0, stream>>>(
        xBf, DIMS, 0, 0, wkBf, DIMS, 0, 0, nullptr,
        nullptr, 0, 0, QKSCALE,
        Kbf, nullptr, DIMS, 0, 0, nullptr, zeroFlags, factor,
        MROWS, DIMS, DIMS);
    // 4) V = x@Wv^T + bv             -> head-transposed bf16
    gemm_wmma_bf16<<<gProj, blk, 0, stream>>>(
        xBf, DIMS, 0, 0, wvwBf, DIMS, 0, 0, bv,
        nullptr, 0, 0, 1.0f,
        nullptr, nullptr, DIMS, 0, 0, vT, nullptr, nullptr,
        MROWS, DIMS, DIMS);

    // 5) qk = (q*s)(k*s)^T * scaled_zero[col]  -> fp32 output, batched per (b,h)
    const dim3 gQK(CTX / 128, CTX / 64, BATCH * HEADS);  // 16 x 32 x 32
    gemm_wmma_bf16<<<gQK, blk, 0, stream>>>(
        Qbf, DIMS, (long long)CTX * DIMS, HD,
        Kbf, DIMS, (long long)CTX * DIMS, HD,
        nullptr,
        zeroFlags, (long long)HEADS * CTX, CTX,
        1.0f,
        nullptr, qkOut, CTX,
        (long long)HEADS * CTX * CTX, (long long)CTX * CTX,
        nullptr, nullptr, nullptr,
        CTX, CTX, HD);

    // 6) softmax row stats
    softmax_stats<<<BATCH * HEADS * CTX, 256, 0, stream>>>(qkOut, rowMax, rowRinv);

    // 7) wv = softmax(qk) @ V  -> bf16 (b, ctx, dims)
    attn_pv_wmma<<<dim3(CTX / 64, 1, BATCH * HEADS), blk, 0, stream>>>(
        qkOut, rowMax, rowRinv, vT, wvBf);

    // 8) out1 = wv@Wout^T + bout -> bf16
    gemm_wmma_bf16<<<gProj, blk, 0, stream>>>(
        wvBf, DIMS, 0, 0, woBf, DIMS, 0, 0, bout,
        nullptr, 0, 0, 1.0f,
        o1Bf, nullptr, DIMS, 0, 0, nullptr, nullptr, nullptr,
        MROWS, DIMS, DIMS);
    // 9) out = out1@Wout^T + bout -> fp32 to d_out
    gemm_wmma_bf16<<<gProj, blk, 0, stream>>>(
        o1Bf, DIMS, 0, 0, woBf, DIMS, 0, 0, bout,
        nullptr, 0, 0, 1.0f,
        nullptr, outPtr, DIMS, 0, 0, nullptr, nullptr, nullptr,
        MROWS, DIMS, DIMS);
}
